// EncoderLayer_22093311770940
// MI455X (gfx1250) — matrix operand order repaired and checked by reading, not compiled, not run on hardware
//
#include <hip/hip_runtime.h>

// ---------------------------------------------------------------- constants
#define LSEQ   2048
#define DMODEL 1024
#define FDIM   4096
#define NBATCH 8
#define MROWS  (NBATCH * LSEQ)   // 16384
#define TOPK   7
#define QPAD   8                 // row padding (halfwords) for LDS bank spread

typedef __bf16 bf16_t;
typedef __attribute__((ext_vector_type(16))) __bf16 v16bf;
typedef __attribute__((ext_vector_type(8)))  float  v8f;

// ---------------------------------------------------------------- helpers
__device__ __forceinline__ unsigned short f2bf(float f) {
    unsigned u = __float_as_uint(f);
    u += 0x7FFFu + ((u >> 16) & 1u);        // round-to-nearest-even
    return (unsigned short)(u >> 16);
}
__device__ __forceinline__ float bf2f(unsigned short h) {
    return __uint_as_float(((unsigned)h) << 16);
}

union FragBF { uint4 u[2]; v16bf v; };

// A operand: lane = M row (lane%16); K pattern {8h+0..7, 16+8h+0..7}
__device__ __forceinline__ v16bf load_a_frag(const unsigned short* base, int hf) {
    FragBF f;
    f.u[0] = *(const uint4*)(base + 8 * hf);
    f.u[1] = *(const uint4*)(base + 16 + 8 * hf);
    return f.v;
}
// B operand: lane = N column (lane%16); K pattern {8h+0..7, 16+8h+0..7}
__device__ __forceinline__ v16bf load_b_frag(const unsigned short* base, int hf) {
    FragBF f;
    f.u[0] = *(const uint4*)(base + 8 * hf);
    f.u[1] = *(const uint4*)(base + 16 + 8 * hf);
    return f.v;
}
__device__ __forceinline__ v8f wmma_bf16(v16bf a, v16bf b, v8f c) {
    return __builtin_amdgcn_wmma_f32_16x16x32_bf16(false, a, false, b,
                                                   (short)0, c, false, false);
}

// Async global -> LDS copy (16B per lane), tracked by ASYNCcnt.
__device__ __forceinline__ void async_copy_b128(unsigned lds_off, const void* gptr) {
    asm volatile("global_load_async_to_lds_b128 %0, %1, off"
                 :: "v"(lds_off), "v"(gptr) : "memory");
}
__device__ __forceinline__ void wait_async0() {
    asm volatile("s_wait_asynccnt 0x0" ::: "memory");
}

// ---------------------------------------------------------------- converts
__global__ void cvt_f32_bf16_kernel(const float* __restrict__ src,
                                    unsigned short* __restrict__ dst, size_t n) {
    size_t i = (size_t)blockIdx.x * blockDim.x + threadIdx.x;
    if (i < n) dst[i] = f2bf(src[i]);
}

// src row-major [R x C] -> dst row-major [C x R] (bf16)
__global__ void transpose_cvt_kernel(const float* __restrict__ src,
                                     unsigned short* __restrict__ dst,
                                     int R, int C) {
    size_t i = (size_t)blockIdx.x * blockDim.x + threadIdx.x;
    size_t n = (size_t)R * C;
    if (i < n) {
        int r = (int)(i / C), c = (int)(i % C);
        dst[(size_t)c * R + r] = f2bf(src[i]);
    }
}

// ---------------------------------------------------------------- GEMM
// C[M,N] = A[M,K](bf16) * Bt[N,K](bf16)^T + bias ; optional ReLU; out f32/bf16
// 256 threads = 8 waves in 2(M) x 4(N); block tile 128x256; wave tile 64x64.
// 16 b128 loads feed 16 WMMAs per K-chunk of 32.
template<bool RELU, bool OUTBF>
__global__ void __launch_bounds__(256)
gemm_bf16_kernel(const unsigned short* __restrict__ A,
                 const unsigned short* __restrict__ Bt,
                 const float* __restrict__ bias,
                 void* __restrict__ out, int M, int N, int K) {
    const int lane = threadIdx.x & 31;
    const int wid  = threadIdx.x >> 5;
    const int hf   = lane >> 4;
    const int r16  = lane & 15;
    const int wm   = wid >> 2;          // 0..1
    const int wn   = wid & 3;           // 0..3
    const int m_base = blockIdx.x * 128 + wm * 64;
    const int n_base = blockIdx.y * 256 + wn * 64;

    v8f acc[4][4];
#pragma unroll
    for (int mt = 0; mt < 4; ++mt)
#pragma unroll
        for (int nt = 0; nt < 4; ++nt)
            acc[mt][nt] = (v8f){0.f, 0.f, 0.f, 0.f, 0.f, 0.f, 0.f, 0.f};

    const unsigned short* arow[4];
#pragma unroll
    for (int mt = 0; mt < 4; ++mt)
        arow[mt] = A + (size_t)(m_base + mt * 16 + r16) * K;
    const unsigned short* brow[4];
#pragma unroll
    for (int nt = 0; nt < 4; ++nt)
        brow[nt] = Bt + (size_t)(n_base + nt * 16 + r16) * K;

    for (int k0 = 0; k0 < K; k0 += 32) {
        v16bf a[4], b[4];
#pragma unroll
        for (int mt = 0; mt < 4; ++mt) a[mt] = load_a_frag(arow[mt] + k0, hf);
#pragma unroll
        for (int nt = 0; nt < 4; ++nt) b[nt] = load_b_frag(brow[nt] + k0, hf);
#pragma unroll
        for (int mt = 0; mt < 4; ++mt)
#pragma unroll
            for (int nt = 0; nt < 4; ++nt)
                acc[mt][nt] = wmma_bf16(a[mt], b[nt], acc[mt][nt]);
    }

#pragma unroll
    for (int nt = 0; nt < 4; ++nt) {
        const int n = n_base + nt * 16 + r16;
        const float bv = bias[n];
#pragma unroll
        for (int mt = 0; mt < 4; ++mt) {
#pragma unroll
            for (int r = 0; r < 8; ++r) {
                int row = m_base + mt * 16 + r + 8 * hf;
                float v = acc[mt][nt][r] + bv;
                if (RELU) v = fmaxf(v, 0.f);
                size_t o = (size_t)row * N + n;
                if (OUTBF) ((unsigned short*)out)[o] = f2bf(v);
                else       ((float*)out)[o] = v;
            }
        }
    }
}

// ---------------------------------------------------------------- correlation
// partial[b, ttile, tau] = sum over rows t in [16*ttile,16*ttile+16), all s of
//   dot(q[b,t,:], k[b,s,:]) binned by tau=(t-s) mod L.
// Q strip (16x1024 bf16, shared by all 8 waves) is staged once into LDS via
// global_load_async_to_lds_b128; A-fragments then come from ds_load_b128 while
// B (k rows, zero reuse) streams from global. Lag histogram via ds_add_f32.
__global__ void __launch_bounds__(256)
corr_partial_kernel(const unsigned short* __restrict__ qb,
                    const unsigned short* __restrict__ kb,
                    float* __restrict__ partial) {
    __shared__ float lag[LSEQ];
    __shared__ __align__(16) unsigned short qs[16][DMODEL + QPAD];

    const int tid = threadIdx.x;
    const int b = blockIdx.y, ttile = blockIdx.x;
    const int t0 = ttile * 16;

    // 1) kick off async staging of the q strip: 2048 chunks of 16B
#pragma unroll
    for (int j = 0; j < 8; ++j) {
        int chunk = tid + j * 256;              // 0..2047
        int row   = chunk >> 7;                 // 0..15
        int col   = (chunk & 127) * 8;          // element offset, 16B granules
        const unsigned short* g =
            qb + ((size_t)b * LSEQ + t0 + row) * DMODEL + col;
        unsigned lds_off = (unsigned)(size_t)(&qs[row][col]);  // low 32b = LDS offset
        async_copy_b128(lds_off, g);
    }

    // 2) zero the lag histogram while the DMA is in flight
    for (int i = tid; i < LSEQ; i += 256) lag[i] = 0.f;

    wait_async0();
    __syncthreads();

    const int lane = tid & 31, wid = tid >> 5;
    const int hf = lane >> 4, r16 = lane & 15;

    for (int st = 0; st < 16; ++st) {
        const int s0 = wid * 256 + st * 16;
        const unsigned short* krow =
            kb + ((size_t)b * LSEQ + s0 + r16) * DMODEL;
        v8f acc = (v8f){0.f, 0.f, 0.f, 0.f, 0.f, 0.f, 0.f, 0.f};
        for (int k0 = 0; k0 < DMODEL; k0 += 32) {
            FragBF fa;
            fa.u[0] = *(const uint4*)(&qs[r16][k0 + 8 * hf]);
            fa.u[1] = *(const uint4*)(&qs[r16][k0 + 16 + 8 * hf]);
            v16bf bf = load_b_frag(krow + k0, hf);
            acc = wmma_bf16(fa.v, bf, acc);
        }
#pragma unroll
        for (int r = 0; r < 8; ++r) {
            int t = t0 + r + 8 * hf;
            int s = s0 + r16;
            int tau = (t - s) & (LSEQ - 1);
            atomicAdd(&lag[tau], acc[r]);
        }
    }
    __syncthreads();
    float* dst = partial + ((size_t)b * 128 + ttile) * LSEQ;
    for (int i = tid; i < LSEQ; i += 256) dst[i] = lag[i];
}

__global__ void corr_reduce_kernel(const float* __restrict__ partial,
                                   float* __restrict__ meancorr) {
    int b = blockIdx.y;
    int tau = blockIdx.x * 256 + threadIdx.x;
    const float* src = partial + (size_t)b * 128 * LSEQ + tau;
    float s = 0.f;
    for (int t = 0; t < 128; ++t) s += src[(size_t)t * LSEQ];
    meancorr[b * LSEQ + tau] = s * (1.0f / DMODEL);   // mean over H*DH
}

// ---------------------------------------------------------------- top-k + softmax
__global__ void __launch_bounds__(256)
topk_softmax_kernel(const float* __restrict__ meancorr,
                    float* __restrict__ wts, int* __restrict__ inds) {
    __shared__ float sv[LSEQ];
    __shared__ float rv[256];
    __shared__ int   ri[256];
    __shared__ float bestv[TOPK];
    __shared__ int   besti[TOPK];
    const int b = blockIdx.x, tid = threadIdx.x;
    for (int i = tid; i < LSEQ; i += 256) sv[i] = meancorr[b * LSEQ + i];
    __syncthreads();

    for (int it = 0; it < TOPK; ++it) {
        float bv = -3.4e38f; int bi = 0;
        for (int i = tid * 8; i < tid * 8 + 8; ++i) {
            float v = sv[i];
            if (v > bv) { bv = v; bi = i; }       // stable: smaller idx wins ties
        }
        rv[tid] = bv; ri[tid] = bi;
        __syncthreads();
        for (int off = 128; off > 0; off >>= 1) {
            if (tid < off) {
                if (rv[tid + off] > rv[tid] ||
                    (rv[tid + off] == rv[tid] && ri[tid + off] < ri[tid])) {
                    rv[tid] = rv[tid + off]; ri[tid] = ri[tid + off];
                }
            }
            __syncthreads();
        }
        if (tid == 0) { bestv[it] = rv[0]; besti[it] = ri[0]; sv[ri[0]] = -3.4e38f; }
        __syncthreads();
    }
    if (tid == 0) {
        float mx = bestv[0];
        float e[TOPK], s = 0.f;
        for (int i = 0; i < TOPK; ++i) { e[i] = __expf(bestv[i] - mx); s += e[i]; }
        for (int i = 0; i < TOPK; ++i) {
            wts[b * TOPK + i]  = e[i] / s;
            inds[b * TOPK + i] = besti[i];
        }
    }
}

// ---------------------------------------------------------------- delay aggregation
// out[b,l,c] = sum_i w[b,i] * v[b,(l+delay_i) mod L, c]; delays from batch 0.
__global__ void __launch_bounds__(128)
aggregate_kernel(const unsigned short* __restrict__ vb,
                 const float* __restrict__ wts, const int* __restrict__ inds,
                 unsigned short* __restrict__ aggb) {
    const int b  = blockIdx.x >> 11;
    const int l  = blockIdx.x & (LSEQ - 1);
    const int c0 = threadIdx.x * 8;
    float acc[8] = {0.f, 0.f, 0.f, 0.f, 0.f, 0.f, 0.f, 0.f};
#pragma unroll
    for (int i = 0; i < TOPK; ++i) {
        const float w = wts[b * TOPK + i];
        const int   d = inds[i];                       // batch-0 delays
        const int row = (l + d) & (LSEQ - 1);
        uint4 src = *(const uint4*)(vb + ((size_t)b * LSEQ + row) * DMODEL + c0);
        const unsigned short* sp = (const unsigned short*)&src;
#pragma unroll
        for (int e = 0; e < 8; ++e) acc[e] += w * bf2f(sp[e]);
    }
    uint4 o; unsigned short* op = (unsigned short*)&o;
#pragma unroll
    for (int e = 0; e < 8; ++e) op[e] = f2bf(acc[e]);
    *(uint4*)(aggb + ((size_t)b * LSEQ + l) * DMODEL + c0) = o;
}

// ---------------------------------------------------------------- LayerNorm(base+add)
__global__ void __launch_bounds__(256)
ln_kernel(const float* __restrict__ base, const float* __restrict__ add,
          const float* __restrict__ g, const float* __restrict__ beta,
          float* __restrict__ outf, unsigned short* __restrict__ outb) {
    __shared__ float red[256];
    const size_t row = blockIdx.x;
    const int tid = threadIdx.x;
    const float* pb = base + row * DMODEL;
    const float* pa = add + row * DMODEL;

    float vals[4], s = 0.f;
#pragma unroll
    for (int k = 0; k < 4; ++k) {
        int c = tid + k * 256;
        vals[k] = pb[c] + pa[c];
        s += vals[k];
    }
    red[tid] = s; __syncthreads();
    for (int off = 128; off > 0; off >>= 1) {
        if (tid < off) red[tid] += red[tid + off];
        __syncthreads();
    }
    const float mean = red[0] * (1.f / DMODEL);
    __syncthreads();

    float sq = 0.f;
#pragma unroll
    for (int k = 0; k < 4; ++k) { float d = vals[k] - mean; sq += d * d; }
    red[tid] = sq; __syncthreads();
    for (int off = 128; off > 0; off >>= 1) {
        if (tid < off) red[tid] += red[tid + off];
        __syncthreads();
    }
    const float var = red[0] * (1.f / DMODEL);
    const float rs = rsqrtf(var + 1e-6f);
#pragma unroll
    for (int k = 0; k < 4; ++k) {
        int c = tid + k * 256;
        float y = (vals[k] - mean) * rs * g[c] + beta[c];
        outf[row * DMODEL + c] = y;
        if (outb) outb[row * DMODEL + c] = f2bf(y);
    }
}

// ---------------------------------------------------------------- launch
extern "C" void kernel_launch(void* const* d_in, const int* in_sizes, int n_in,
                              void* d_out, int out_size, void* d_ws, size_t ws_size,
                              hipStream_t stream) {
    (void)in_sizes; (void)n_in; (void)out_size; (void)ws_size;

    const float* x    = (const float*)d_in[0];
    const float* Wq   = (const float*)d_in[1];
    const float* bq   = (const float*)d_in[2];
    const float* Wk   = (const float*)d_in[3];
    const float* bk   = (const float*)d_in[4];
    const float* Wv   = (const float*)d_in[5];
    const float* bv   = (const float*)d_in[6];
    const float* Wo   = (const float*)d_in[7];
    const float* bo   = (const float*)d_in[8];
    const float* ln1g = (const float*)d_in[9];
    const float* ln1b = (const float*)d_in[10];
    const float* c1w  = (const float*)d_in[11];
    const float* c1b  = (const float*)d_in[12];
    const float* c2w  = (const float*)d_in[13];
    const float* c2b  = (const float*)d_in[14];
    const float* ln2g = (const float*)d_in[15];
    const float* ln2b = (const float*)d_in[16];

    char* ws = (char*)d_ws;
    size_t off = 0;
    auto alloc = [&](size_t bytes) -> char* {
        char* p = ws + off;
        off += (bytes + 255) & ~(size_t)255;
        return p;
    };

    const size_t szMD2 = (size_t)MROWS * DMODEL * 2;   // 33.5 MB (bf16 M x D)

    unsigned short* wqT  = (unsigned short*)alloc((size_t)DMODEL * DMODEL * 2);
    unsigned short* wkT  = (unsigned short*)alloc((size_t)DMODEL * DMODEL * 2);
    unsigned short* wvT  = (unsigned short*)alloc((size_t)DMODEL * DMODEL * 2);
    unsigned short* woT  = (unsigned short*)alloc((size_t)DMODEL * DMODEL * 2);
    unsigned short* c1T  = (unsigned short*)alloc((size_t)DMODEL * FDIM * 2);
    unsigned short* c2T  = (unsigned short*)alloc((size_t)FDIM * DMODEL * 2);
    float* meancorr      = (float*)alloc((size_t)NBATCH * LSEQ * 4);
    float* wts           = (float*)alloc(NBATCH * TOPK * 4);
    int*   inds          = (int*)alloc(NBATCH * TOPK * 4);
    float* partial       = (float*)alloc((size_t)NBATCH * 128 * LSEQ * 4);
    unsigned short* xb   = (unsigned short*)alloc(szMD2);
    unsigned short* aggb = (unsigned short*)alloc(szMD2);   // adjacent to xb
    unsigned short* qb   = (unsigned short*)alloc(szMD2);
    unsigned short* kb   = (unsigned short*)alloc(szMD2);   // adjacent to qb
    unsigned short* vb   = (unsigned short*)alloc(szMD2);
    float* x1f           = (float*)alloc((size_t)MROWS * DMODEL * 4);
    unsigned short* hb   = (unsigned short*)alloc((size_t)MROWS * FDIM * 2);

    // aliases (lifetime-disjoint reuse)
    float* attn          = (float*)qb;          // spans qb+kb   (dead after corr)
    unsigned short* x1b  = vb;                  // vb dead after aggregate
    float* yf            = (float*)xb;          // spans xb+aggb (dead after gemms)

    // 1) precision conversion + weight transposes
    const size_t nx = (size_t)MROWS * DMODEL;
    cvt_f32_bf16_kernel<<<(unsigned)((nx + 255) / 256), 256, 0, stream>>>(x, xb, nx);
    transpose_cvt_kernel<<<(DMODEL * DMODEL + 255) / 256, 256, 0, stream>>>(Wq, wqT, DMODEL, DMODEL);
    transpose_cvt_kernel<<<(DMODEL * DMODEL + 255) / 256, 256, 0, stream>>>(Wk, wkT, DMODEL, DMODEL);
    transpose_cvt_kernel<<<(DMODEL * DMODEL + 255) / 256, 256, 0, stream>>>(Wv, wvT, DMODEL, DMODEL);
    transpose_cvt_kernel<<<(DMODEL * DMODEL + 255) / 256, 256, 0, stream>>>(Wo, woT, DMODEL, DMODEL);
    transpose_cvt_kernel<<<(DMODEL * FDIM + 255) / 256, 256, 0, stream>>>(c1w, c1T, DMODEL, FDIM);
    transpose_cvt_kernel<<<(FDIM * DMODEL + 255) / 256, 256, 0, stream>>>(c2w, c2T, FDIM, DMODEL);

    // 2) Q/K/V projections (WMMA bf16 -> bf16)
    dim3 gD(MROWS / 128, DMODEL / 256);
    gemm_bf16_kernel<false, true><<<gD, 256, 0, stream>>>(xb, wqT, bq, qb, MROWS, DMODEL, DMODEL);
    gemm_bf16_kernel<false, true><<<gD, 256, 0, stream>>>(xb, wkT, bk, kb, MROWS, DMODEL, DMODEL);
    gemm_bf16_kernel<false, true><<<gD, 256, 0, stream>>>(xb, wvT, bv, vb, MROWS, DMODEL, DMODEL);

    // 3) auto-correlation lag scores via Q*K^T diagonal binning (WMMA + async LDS)
    corr_partial_kernel<<<dim3(128, NBATCH), 256, 0, stream>>>(qb, kb, partial);
    corr_reduce_kernel<<<dim3(LSEQ / 256, NBATCH), 256, 0, stream>>>(partial, meancorr);

    // 4) top-7 lags + softmax weights
    topk_softmax_kernel<<<NBATCH, 256, 0, stream>>>(meancorr, wts, inds);

    // 5) weighted circular-roll aggregation of V
    aggregate_kernel<<<NBATCH * LSEQ, 128, 0, stream>>>(vb, wts, inds, aggb);

    // 6) output projection, residual + LN1
    gemm_bf16_kernel<false, false><<<gD, 256, 0, stream>>>(aggb, woT, bo, attn, MROWS, DMODEL, DMODEL);
    ln_kernel<<<MROWS, 256, 0, stream>>>(x, attn, ln1g, ln1b, x1f, x1b);

    // 7) FFN: relu(x1 @ c1 + b1) @ c2 + b2, residual + LN2
    dim3 gF(MROWS / 128, FDIM / 256);
    gemm_bf16_kernel<true, true><<<gF, 256, 0, stream>>>(x1b, c1T, c1b, hb, MROWS, FDIM, DMODEL);
    gemm_bf16_kernel<false, false><<<gD, 256, 0, stream>>>(hb, c2T, c2b, yf, MROWS, DMODEL, FDIM);
    ln_kernel<<<MROWS, 256, 0, stream>>>(x1f, yf, ln2g, ln2b, (float*)d_out, nullptr);
}
